// SpatialTransformer_5257039970565
// MI455X (gfx1250) — compile-verified, matrix operand order checked by reading, not run on hardware
//
#include <hip/hip_runtime.h>
#include <hip/hip_bf16.h>

typedef __attribute__((ext_vector_type(16))) _Float16 v16h;
typedef __attribute__((ext_vector_type(8)))  _Float16 v8h;
typedef __attribute__((ext_vector_type(8)))  float    v8f;

constexpr int kB  = 2;
constexpr int kN  = 1024;
constexpr int kH  = 512;
constexpr int kNH = 8;
constexpr int kHD = 64;
constexpr int kG  = 50;

// ---------------------------------------------------------------------------
// A-fragment load (CDNA5 16-bit 16x32 A layout, ISA 7.12.2).
// Lane l: row m = l&15, half = l>>4. Per-lane data = two contiguous 16B
// chunks: K = half*8 + [0..7] and K = half*8 + [16..23]  ->  2x b128 loads.
// ---------------------------------------------------------------------------
__device__ __forceinline__ v16h load_frag_a(const _Float16* __restrict__ p, int ld, int lane) {
  int m = lane & 15, half = lane >> 4;
  const _Float16* base = p + (size_t)m * ld + half * 8;
  v8h c0 = *(const v8h*)(base);
  v8h c1 = *(const v8h*)(base + 16);
  v16h f;
#pragma unroll
  for (int i = 0; i < 8; ++i) { f[i] = c0[i]; f[8 + i] = c1[i]; }
  return f;
}

// ---------------------------------------------------------------------------
// DPP16 row_ror reductions within 16-lane rows (the C/D half-groups).
// Pure VALU — replaces ds_bpermute + s_wait_dscnt round-trips.
// ---------------------------------------------------------------------------
template <int RN>
__device__ __forceinline__ float dpp_ror(float x) {
  return __builtin_bit_cast(float,
      __builtin_amdgcn_update_dpp(0, __builtin_bit_cast(int, x),
                                  0x120 + RN, 0xF, 0xF, true));
}
__device__ __forceinline__ float rowmax16(float x) {
  x = fmaxf(x, dpp_ror<1>(x));
  x = fmaxf(x, dpp_ror<2>(x));
  x = fmaxf(x, dpp_ror<4>(x));
  x = fmaxf(x, dpp_ror<8>(x));
  return x;
}
__device__ __forceinline__ float rowsum16(float x) {
  x += dpp_ror<1>(x);
  x += dpp_ror<2>(x);
  x += dpp_ror<4>(x);
  x += dpp_ror<8>(x);
  return x;
}

// ---------------------------------------------------------------------------
// Pack a row-major (K x Nn) matrix into B-fragment order:
// dst[((kt*(Nn/16)+nt)*32 + lane)*16 + e] = src[k][n]
// with n = nt*16 + (lane&15),
//      k = kt*32 + ((e>>1)&3)*2 + ((e>>3)<<4) + ((lane>>4)<<3) + (e&1)
// Consumers then fetch one v16h per lane per fragment.
// ---------------------------------------------------------------------------
template <typename T>
__global__ void pack_bfrag_kernel(const T* __restrict__ src, _Float16* __restrict__ dst,
                                  int K, int Nn) {
  int idx = blockIdx.x * 256 + threadIdx.x;
  if (idx >= K * Nn) return;
  int e    = idx & 15;
  int lane = (idx >> 4) & 31;
  int frag = idx >> 9;
  int nfragn = Nn >> 4;
  int kt = frag / nfragn, nt = frag - kt * nfragn;
  int v = e >> 1;
  int k = kt * 32 + ((v & 3) << 1) + ((v >> 2) << 4) + ((lane >> 4) << 3) + (e & 1);
  int n = (nt << 4) + (lane & 15);
  dst[idx] = (_Float16)src[(size_t)k * Nn + n];
}

// ---------------------------------------------------------------------------
// LayerNorm over H=512, one block (128 threads) per row, f16 output
// ---------------------------------------------------------------------------
__global__ __launch_bounds__(128)
void layernorm_kernel(const float* __restrict__ x, const float* __restrict__ g,
                      const float* __restrict__ bt, _Float16* __restrict__ out) {
  __shared__ float red[128];
  int row = blockIdx.x, t = threadIdx.x;
  const float* xr = x + (size_t)row * kH;
  float lv[4];
  float s = 0.f;
#pragma unroll
  for (int i = 0; i < 4; ++i) { lv[i] = xr[t + i * 128]; s += lv[i]; }
  red[t] = s; __syncthreads();
  for (int off = 64; off > 0; off >>= 1) { if (t < off) red[t] += red[t + off]; __syncthreads(); }
  float mu = red[0] * (1.f / kH);
  __syncthreads();
  float q = 0.f;
#pragma unroll
  for (int i = 0; i < 4; ++i) { float d = lv[i] - mu; q += d * d; }
  red[t] = q; __syncthreads();
  for (int off = 64; off > 0; off >>= 1) { if (t < off) red[t] += red[t + off]; __syncthreads(); }
  float rstd = rsqrtf(red[0] * (1.f / kH) + 1e-5f);
#pragma unroll
  for (int i = 0; i < 4; ++i) {
    int c = t + i * 128;
    out[(size_t)row * kH + c] = (_Float16)((lv[i] - mu) * rstd * g[c] + bt[c]);
  }
}

// ---------------------------------------------------------------------------
// Distance bias, written directly in D-fragment-packed order:
// biasP[frag*256 + lane*8 + r] for frag = ((b*8+h)*64 + q/16)*64 + k/16,
// lane = (k&15) + 16*((q>>3)&1), r = q&7. Attention reads one b128/lane/frag.
// ---------------------------------------------------------------------------
__global__ __launch_bounds__(256)
void dist_bias_kernel(const float* __restrict__ dist, const float* __restrict__ widths,
                      const float* __restrict__ dp_w, const float* __restrict__ dp_b,
                      _Float16* __restrict__ biasP) {
  __shared__ float s_inv[kG];
  __shared__ float s_w[kG * kNH];
  int tid = threadIdx.x;
  if (tid < kG) { float wv = widths[tid]; s_inv[tid] = 1.f / (2.f * wv * wv); }
  for (int i = tid; i < kG * kNH; i += 256) s_w[i] = dp_w[i];
  __syncthreads();

  int idx = blockIdx.x * 256 + tid;             // (b*N + q)*N + k, exact grid
  int k = idx & (kN - 1);
  int q = (idx >> 10) & (kN - 1);
  int b = idx >> 20;
  float d = dist[idx];
  float acch[kNH];
#pragma unroll
  for (int h = 0; h < kNH; ++h) acch[h] = dp_b[h];
  const float step = 10.f / 49.f;               // linspace(0,10,50)
  for (int g = 0; g < kG; ++g) {
    float dd = d - g * step;
    float e = __expf(-dd * dd * s_inv[g]);
#pragma unroll
    for (int h = 0; h < kNH; ++h) acch[h] += e * s_w[g * kNH + h];
  }
  int lane = (k & 15) + (((q >> 3) & 1) << 4);
  int r = q & 7;
  size_t fragRow = ((size_t)(q >> 4) * 64 + (k >> 4)) * 256 + (size_t)lane * 8 + r;
#pragma unroll
  for (int h = 0; h < kNH; ++h)
    biasP[(size_t)(b * kNH + h) * (64 * 64 * 256) + fragRow] = (_Float16)acch[h];
}

// ---------------------------------------------------------------------------
// WMMA GEMM: C[M,Nn] = A[M,K] @ W[K,Nn] (+bias), W pre-packed in B-frag order.
// Block = 128 threads = 4 waves; wave owns 32 rows of a 128x64 block tile
// (2 A-frags amortize each B-frag -> 8 wmma per K-step per wave).
// EPI 0: outH = f16(val*scale) | EPI 1: f16(gelu) | EPI 2: outF = residual+val
// ---------------------------------------------------------------------------
template <int EPI>
__global__ __launch_bounds__(128)
void gemm_wmma(const _Float16* __restrict__ A, const _Float16* __restrict__ Wpk,
               const float* __restrict__ bias, const float* __restrict__ residual,
               _Float16* __restrict__ outH, float* __restrict__ outF,
               int K, int Nn, float scale) {
  int lane = threadIdx.x & 31;
  int wave = threadIdx.x >> 5;
  int row0 = blockIdx.y * 128 + wave * 32;
  int col0 = blockIdx.x * 64;
  int half = lane >> 4, nl = lane & 15;
  int nfragn = Nn >> 4;
  const v16h* Wp = (const v16h*)Wpk;

  v8f acc[2][4] = {};
  for (int k0 = 0; k0 < K; k0 += 32) {
    v16h a0 = load_frag_a(A + (size_t)row0 * K + k0, K, lane);
    v16h a1 = load_frag_a(A + (size_t)(row0 + 16) * K + k0, K, lane);
    if (k0 + 32 < K) __builtin_prefetch(A + (size_t)row0 * K + k0 + 32, 0, 0);
    const v16h* wb = Wp + ((size_t)(k0 >> 5) * nfragn + (col0 >> 4)) * 32 + lane;
#pragma unroll
    for (int t = 0; t < 4; ++t) {
      v16h bf = wb[(size_t)t * 32];
      acc[0][t] = __builtin_amdgcn_wmma_f32_16x16x32_f16(false, a0, false, bf,
                                                         (short)0, acc[0][t], false, false);
      acc[1][t] = __builtin_amdgcn_wmma_f32_16x16x32_f16(false, a1, false, bf,
                                                         (short)0, acc[1][t], false, false);
    }
  }

#pragma unroll
  for (int m = 0; m < 2; ++m) {
#pragma unroll
    for (int t = 0; t < 4; ++t) {
      int col = col0 + t * 16 + nl;
      float bv = bias[col];
#pragma unroll
      for (int r = 0; r < 8; ++r) {
        int row = row0 + m * 16 + r + 8 * half;
        float v = acc[m][t][r] + bv;
        size_t idx = (size_t)row * Nn + col;
        if (EPI == 0) {
          outH[idx] = (_Float16)(v * scale);
        } else if (EPI == 1) {
          float gl = 0.5f * v * (1.f + erff(v * 0.70710678118654752f));
          outH[idx] = (_Float16)gl;
        } else {
          outF[idx] = residual[idx] + v;
        }
      }
    }
  }
}

// ---------------------------------------------------------------------------
// Flash attention: one wave per (b, h, 16-row q tile). Streams 32-key tiles:
//   S = Q Kt (4 wmma) -> +packed bias/mask -> online softmax (DPP row_ror)
//   -> P via LDS transpose -> out += P V (4 wmma, V pre-packed B-frags).
// ---------------------------------------------------------------------------
__global__ __launch_bounds__(128)
void attention_kernel(const _Float16* __restrict__ Qh, const _Float16* __restrict__ Kh,
                      const _Float16* __restrict__ Vpk, const _Float16* __restrict__ biasP,
                      const int* __restrict__ mask, _Float16* __restrict__ outH) {
  __shared__ __align__(16) _Float16 plds[4][16 * 32];
  int lane = threadIdx.x & 31;
  int wib  = threadIdx.x >> 5;
  int w    = blockIdx.x * 4 + wib;
  int qt = w & 63;
  int h  = (w >> 6) & 7;
  int b  = w >> 9;
  int q0 = qt << 4;
  int half = lane >> 4, nl = lane & 15;

  const _Float16* Qp = Qh + (size_t)b * kN * kH + (size_t)h * kHD;
  const _Float16* Kp = Kh + (size_t)b * kN * kH + (size_t)h * kHD;
  const v16h*     Vp = (const v16h*)Vpk;          // packed over rows = b*kN + key
  const _Float16* Bp = biasP + (size_t)(b * kNH + h) * (64 * 64 * 256)
                             + (size_t)qt * (64 * 256);
  const int* mp = mask + (size_t)b * kN;

  v16h a0 = load_frag_a(Qp + (size_t)q0 * kH, kH, lane);       // Q pre-scaled 1/8
  v16h a1 = load_frag_a(Qp + (size_t)q0 * kH + 32, kH, lane);

  float mrow[8], lrow[8];
#pragma unroll
  for (int r = 0; r < 8; ++r) { mrow[r] = -1e30f; lrow[r] = 0.f; }
  v8f acc[4] = {};

  for (int kt = 0; kt < kN / 32; ++kt) {
    int key0 = kt * 32;
    v8f s[2] = {};
#pragma unroll
    for (int t = 0; t < 2; ++t) {
      // Kt fragment == A-layout load of K rows (keys) x d columns
      v16h k0f = load_frag_a(Kp + (size_t)(key0 + t * 16) * kH, kH, lane);
      v16h k1f = load_frag_a(Kp + (size_t)(key0 + t * 16) * kH + 32, kH, lane);
      s[t] = __builtin_amdgcn_wmma_f32_16x16x32_f16(false, a0, false, k0f, (short)0, s[t], false, false);
      s[t] = __builtin_amdgcn_wmma_f32_16x16x32_f16(false, a1, false, k1f, (short)0, s[t], false, false);
    }

    // packed bias (one b128 per lane per 16-key tile) + mask + tile row-max
    float tm[8];
#pragma unroll
    for (int r = 0; r < 8; ++r) tm[r] = -1e30f;
#pragma unroll
    for (int t = 0; t < 2; ++t) {
      int key = key0 + t * 16 + nl;
      float madd = (mp[key] == 0) ? -1e9f : 0.f;
      v8h bfr = *(const v8h*)(Bp + (size_t)((key0 >> 4) + t) * 256 + (size_t)lane * 8);
#pragma unroll
      for (int r = 0; r < 8; ++r) {
        float sv = s[t][r] + (float)bfr[r] + madd;
        s[t][r] = sv;
        tm[r] = fmaxf(tm[r], sv);
      }
    }
#pragma unroll
    for (int r = 0; r < 8; ++r) tm[r] = rowmax16(tm[r]);

    // online rescale of running stats + accumulators
    float rs[8];
#pragma unroll
    for (int r = 0; r < 8; ++r) {
      float mn = fmaxf(mrow[r], tm[r]);
      float sc = __expf(mrow[r] - mn);
      mrow[r] = mn;
      lrow[r] *= sc;
#pragma unroll
      for (int t = 0; t < 4; ++t) acc[t][r] *= sc;
      rs[r] = 0.f;
    }

    // P = exp(S - m): stage to LDS row-major for A-fragment reload
#pragma unroll
    for (int t = 0; t < 2; ++t) {
#pragma unroll
      for (int r = 0; r < 8; ++r) {
        float p = __expf(s[t][r] - mrow[r]);
        rs[r] += p;
        plds[wib][(r + 8 * half) * 32 + t * 16 + nl] = (_Float16)p;
      }
    }
#pragma unroll
    for (int r = 0; r < 8; ++r) lrow[r] += rowsum16(rs[r]);

    asm volatile("s_wait_dscnt 0" ::: "memory");   // order LDS transpose (intra-wave)
    v16h pfrag = load_frag_a(&plds[wib][0], 32, lane);
#pragma unroll
    for (int t = 0; t < 4; ++t) {
      v16h vf = Vp[((size_t)((b * kN + key0) >> 5) * (kH >> 4) + (h * 4 + t)) * 32 + lane];
      acc[t] = __builtin_amdgcn_wmma_f32_16x16x32_f16(false, pfrag, false, vf,
                                                      (short)0, acc[t], false, false);
    }
  }

#pragma unroll
  for (int t = 0; t < 4; ++t) {
#pragma unroll
    for (int r = 0; r < 8; ++r) {
      int qr = q0 + r + 8 * half;
      float o = acc[t][r] / lrow[r];
      outH[((size_t)b * kN + qr) * kH + h * kHD + t * 16 + nl] = (_Float16)o;
    }
  }
}

// ---------------------------------------------------------------------------
extern "C" void kernel_launch(void* const* d_in, const int* in_sizes, int n_in,
                              void* d_out, int out_size, void* d_ws, size_t ws_size,
                              hipStream_t stream) {
  const float* x     = (const float*)d_in[0];
  const float* dist  = (const float*)d_in[1];
  const int*   mask  = (const int*)d_in[2];
  const float* wdth  = (const float*)d_in[3];
  const float* dp_w  = (const float*)d_in[4];
  const float* dp_b  = (const float*)d_in[5];
  const float* wq    = (const float*)d_in[6];
  const float* bq    = (const float*)d_in[7];
  const float* wk    = (const float*)d_in[8];
  const float* bk    = (const float*)d_in[9];
  const float* wv    = (const float*)d_in[10];
  const float* bv    = (const float*)d_in[11];
  const float* wo    = (const float*)d_in[12];
  const float* bo    = (const float*)d_in[13];
  const float* ln1g  = (const float*)d_in[14];
  const float* ln1b  = (const float*)d_in[15];
  const float* ln2g  = (const float*)d_in[16];
  const float* ln2b  = (const float*)d_in[17];
  const float* w1    = (const float*)d_in[18];
  const float* b1    = (const float*)d_in[19];
  const float* w2    = (const float*)d_in[20];
  const float* b2    = (const float*)d_in[21];

  char* ws = (char*)d_ws;
  size_t off = 0;
  auto alloc = [&](size_t bytes) { char* p = ws + off; off += (bytes + 255) & ~size_t(255); return p; };
  _Float16* wqp  = (_Float16*)alloc((size_t)kH * kH * 2);
  _Float16* wkp  = (_Float16*)alloc((size_t)kH * kH * 2);
  _Float16* wvp  = (_Float16*)alloc((size_t)kH * kH * 2);
  _Float16* wop  = (_Float16*)alloc((size_t)kH * kH * 2);
  _Float16* w1p  = (_Float16*)alloc((size_t)kH * 4 * kH * 2);
  _Float16* w2p  = (_Float16*)alloc((size_t)4 * kH * kH * 2);
  _Float16* xn1  = (_Float16*)alloc((size_t)kB * kN * kH * 2);
  _Float16* qh   = (_Float16*)alloc((size_t)kB * kN * kH * 2);
  _Float16* kh   = (_Float16*)alloc((size_t)kB * kN * kH * 2);
  _Float16* vh   = (_Float16*)alloc((size_t)kB * kN * kH * 2);
  _Float16* vpk  = (_Float16*)alloc((size_t)kB * kN * kH * 2);
  _Float16* biasP= (_Float16*)alloc((size_t)kB * kNH * kN * kN * 2);
  _Float16* atth = (_Float16*)alloc((size_t)kB * kN * kH * 2);
  float*    x1   = (float*)   alloc((size_t)kB * kN * kH * 4);
  _Float16* xn2  = (_Float16*)alloc((size_t)kB * kN * kH * 2);
  _Float16* ffnh = (_Float16*)alloc((size_t)kB * kN * 4 * kH * 2);

  auto packW = [&](const float* src, _Float16* dst, int K, int Nn) {
    pack_bfrag_kernel<float><<<(K * Nn + 255) / 256, 256, 0, stream>>>(src, dst, K, Nn);
  };
  packW(wq, wqp, kH, kH);
  packW(wk, wkp, kH, kH);
  packW(wv, wvp, kH, kH);
  packW(wo, wop, kH, kH);
  packW(w1, w1p, kH, 4 * kH);
  packW(w2, w2p, 4 * kH, kH);

  layernorm_kernel<<<kB * kN, 128, 0, stream>>>(x, ln1g, ln1b, xn1);
  dist_bias_kernel<<<(kB * kN * kN) / 256, 256, 0, stream>>>(dist, wdth, dp_w, dp_b, biasP);

  dim3 g512(kH / 64, (kB * kN) / 128);
  gemm_wmma<0><<<g512, 128, 0, stream>>>(xn1, wqp, bq, nullptr, qh, nullptr, kH, kH, 0.125f);
  gemm_wmma<0><<<g512, 128, 0, stream>>>(xn1, wkp, bk, nullptr, kh, nullptr, kH, kH, 1.0f);
  gemm_wmma<0><<<g512, 128, 0, stream>>>(xn1, wvp, bv, nullptr, vh, nullptr, kH, kH, 1.0f);

  // repack V (rows = b*kN + key) into B-fragment order for attention's P@V
  pack_bfrag_kernel<_Float16><<<(kB * kN * kH + 255) / 256, 256, 0, stream>>>(
      vh, vpk, kB * kN, kH);

  attention_kernel<<<(kB * kNH * (kN / 16)) / 4, 128, 0, stream>>>(qh, kh, vpk, biasP, mask, atth);

  gemm_wmma<2><<<g512, 128, 0, stream>>>(atth, wop, bo, x, nullptr, x1, kH, kH, 1.0f);
  layernorm_kernel<<<kB * kN, 128, 0, stream>>>(x1, ln2g, ln2b, xn2);

  dim3 gffn1((4 * kH) / 64, (kB * kN) / 128);
  gemm_wmma<1><<<gffn1, 128, 0, stream>>>(xn2, w1p, b1, nullptr, ffnh, nullptr, kH, 4 * kH, 1.0f);
  gemm_wmma<2><<<g512, 128, 0, stream>>>(ffnh, w2p, b2, x1, nullptr, (float*)d_out, 4 * kH, kH, 1.0f);

  (void)in_sizes; (void)n_in; (void)out_size; (void)ws_size;
}
